// Decoder_27084063769194
// MI455X (gfx1250) — matrix operand order repaired and checked by reading, not compiled, not run on hardware
//
#include <hip/hip_runtime.h>
#include <hip/hip_bf16.h>

// MI455X / gfx1250, wave32.
// - logits GEMM: v_wmma_f32_16x16x32_bf16, B staged to LDS by the Tensor Data
//   Mover (double-buffered, TENSORcnt-synced), fused masked log_softmax.
// - decoder: one WG (512 thr) per sequence, fp16-packed weights streamed from
//   L2 (halved bytes vs f32), LDS-resident h/c/prefix-sum state.

typedef __attribute__((ext_vector_type(16))) __bf16   v16bf;
typedef __attribute__((ext_vector_type(8)))  float    v8f;
typedef __attribute__((ext_vector_type(8)))  _Float16 h8;
typedef __attribute__((ext_vector_type(4)))  unsigned int v4u;
typedef __attribute__((ext_vector_type(8)))  int      v8i_;
typedef __attribute__((ext_vector_type(4)))  int      v4i_;

#define B_   16
#define T_   2048
#define D_   1024      // 2*Hr
#define L_   120
#define NT_  8         // 8 n-tiles of 16 -> 128 padded cols
#define HR_  512
#define HIN_ 512
#define PD_  64
#define CIN_ 1088      // D_ + PD_

// d_ws layout (16-bit elements)
#define WSOFF_B     0           // 131072 bf16  (packed W_lin fragments, 256 KB)
#define WSOFF_COMB  131072      // 557056 f16   (W_comb)
#define WSOFF_IH    688128      // 1048576 f16  (W_ih)
#define WSOFF_HH    1736704     // 1048576 f16  (W_hh)

__device__ __forceinline__ float sigf(float x) { return 1.0f / (1.0f + __expf(-x)); }

#if defined(__has_builtin)
#if __has_builtin(__builtin_amdgcn_tensor_load_to_lds) && __has_builtin(__builtin_amdgcn_s_wait_tensorcnt)
#define HAVE_TDM 1
#endif
#endif

#ifdef HAVE_TDM
// Issue a TDM copy of one contiguous 8 KB slab (global -> LDS).
// D# built per CDNA5 ISA 8.3/8.4: group0 {count=1, lds_addr, global_addr,
// type=2}; group1 {data_size=8B, tensor_dim0=1024, tile_dim0=1024, 1-D}.
__device__ __forceinline__ void tdm_load_slab(const __bf16* gsrc, unsigned lds_off) {
  unsigned long long ga = (unsigned long long)(uintptr_t)gsrc;
  v4u g0;
  g0[0] = 1u;                                           // count=1 (valid), user mode
  g0[1] = lds_off;                                      // LDS byte address
  g0[2] = (unsigned)(ga & 0xffffffffu);                 // global_addr[31:0]
  g0[3] = (unsigned)((ga >> 32) & 0x01ffffffu) | (2u << 30);  // addr[56:32] | type=2
  v8i_ g1;
  g1[0] = 0x00030000;          // data_size=3 (8B elements); no multicast/pad/iter
  g1[1] = (int)(1024u << 16);  // tensor_dim0 = 1024 (bits 79:48, low half)
  g1[2] = (int)(1u << 16);     // tensor_dim0 hi | tensor_dim1 = 1
  g1[3] = (int)(1024u << 16);  // tensor_dim1 hi | tile_dim0 = 1024
  g1[4] = 1;                   // tile_dim1 = 1, tile_dim2 = 0
  g1[5] = 1024;                // tensor_dim0_stride = 1024 (8B units)
  g1[6] = (int)(1024u << 16);  // stride0 hi | tensor_dim1_stride lo
  g1[7] = 0;
  v4i_ z4 = {};
#if __clang_major__ >= 23
  v8i_ z8 = {};
  __builtin_amdgcn_tensor_load_to_lds(g0, g1, z4, z4, z8, 0);
#else
  __builtin_amdgcn_tensor_load_to_lds(g0, g1, z4, z4, 0);
#endif
}
#endif

// ---------------------------------------------------------------------------
// Kernel 0a: pack W_lin (f32 [120,1024], zero-pad to 128 cols) into bf16 WMMA
// B-fragment order: group g = (kc*8 + tile)*32 + lane holds that lane's 16
// K-contiguous bf16, so a fragment is one 32 B read.
// ---------------------------------------------------------------------------
__global__ void pack_wlin_kernel(const float* __restrict__ W_lin,
                                 __bf16* __restrict__ wsB) {
  int gid = blockIdx.x * blockDim.x + threadIdx.x;   // 32*8*32 = 8192 groups
  int kc  = gid >> 8;
  int rem = gid & 255;
  int t   = rem >> 5;
  int l   = rem & 31;
  int n     = t * 16 + (l & 15);
  int kbase = kc * 32 + ((l < 16) ? 0 : 16);         // B 32x16: lanes>=16 hold K 16..31
  __bf16* dst = wsB + (size_t)gid * 16;
  #pragma unroll
  for (int j = 0; j < 16; ++j) {
    float v = (n < L_) ? W_lin[(size_t)n * D_ + kbase + j] : 0.0f;
    dst[j] = (__bf16)v;
  }
}

// Kernel 0b: generic f32 -> f16 weight pack (grid-stride).
__global__ void pack_f16_kernel(const float* __restrict__ src,
                                _Float16* __restrict__ dst, int n) {
  int stride = gridDim.x * blockDim.x;
  for (int i = blockIdx.x * blockDim.x + threadIdx.x; i < n; i += stride)
    dst[i] = (_Float16)src[i];
}

// ---------------------------------------------------------------------------
// Kernel 1: logits = enc @ W_lin^T (WMMA bf16, f32 accum) + fused masked
// log_softmax. 4 waves/WG, one 16-row M-tile per wave, 8 N-tiles resident.
// B slab (8 KB per K-chunk) is TDM-prefetched into double-buffered LDS.
// ---------------------------------------------------------------------------
__global__ __launch_bounds__(128) void logits_lsm_kernel(
    const float* __restrict__ enc, const __bf16* __restrict__ wsB,
    const int* __restrict__ lengths, float* __restrict__ out) {
  __shared__ __align__(32) __bf16 shB[2][4096];      // 2 x 8 KB B slabs

  const int lane   = threadIdx.x & 31;
  const int wave   = threadIdx.x >> 5;
  const int m_base = blockIdx.x * 64 + wave * 16;
  const int lc     = lane & 15;
  const int koff   = (lane < 16) ? 0 : 16;           // A 16x32 lane K-offset
  const int mrow   = m_base + lc;

  v8f acc[NT_];
  #pragma unroll
  for (int t = 0; t < NT_; ++t) { v8f z = {}; acc[t] = z; }

#ifdef HAVE_TDM
  if (threadIdx.x < 32)                               // wave 0 drives the TDM
    tdm_load_slab(wsB, (unsigned)(uintptr_t)&shB[0][0]);
#endif

  const float* arow = enc + (size_t)mrow * D_;
  for (int kc = 0; kc < D_ / 32; ++kc) {
    const int k0 = kc * 32;
#ifdef HAVE_TDM
    if (threadIdx.x < 32) {
      if (kc + 1 < D_ / 32) {                         // prefetch next slab
        tdm_load_slab(wsB + (size_t)(kc + 1) * 4096,
                      (unsigned)(uintptr_t)&shB[(kc + 1) & 1][0]);
        __builtin_amdgcn_s_wait_tensorcnt(1);         // slab kc landed
      } else {
        __builtin_amdgcn_s_wait_tensorcnt(0);
      }
    }
#else
    {                                                 // cooperative fallback
      const uint4* src = (const uint4*)(wsB + (size_t)kc * 4096);
      uint4*       dst = (uint4*)&shB[kc & 1][0];
      for (int i = threadIdx.x; i < 512; i += 128) dst[i] = src[i];
    }
#endif
    __syncthreads();                                  // slab kc visible to all

    // A fragment: K = k0+koff+0..15 (contiguous, matches packed B)
    const float4* p0 = (const float4*)(arow + k0 + koff);
    const float4* p1 = (const float4*)(arow + k0 + koff + 8);
    float4 a0 = p0[0], a1 = p0[1], a2 = p1[0], a3 = p1[1];
    v16bf a;
    a[0]  = (__bf16)a0.x; a[1]  = (__bf16)a0.y; a[2]  = (__bf16)a0.z; a[3]  = (__bf16)a0.w;
    a[4]  = (__bf16)a1.x; a[5]  = (__bf16)a1.y; a[6]  = (__bf16)a1.z; a[7]  = (__bf16)a1.w;
    a[8]  = (__bf16)a2.x; a[9]  = (__bf16)a2.y; a[10] = (__bf16)a2.z; a[11] = (__bf16)a2.w;
    a[12] = (__bf16)a3.x; a[13] = (__bf16)a3.y; a[14] = (__bf16)a3.z; a[15] = (__bf16)a3.w;

    const v16bf* bb = (const v16bf*)&shB[kc & 1][0];
    #pragma unroll
    for (int t = 0; t < NT_; ++t) {
      v16bf bfrag = bb[t * 32 + lane];                // ds_load_b128 x2
      acc[t] = __builtin_amdgcn_wmma_f32_16x16x32_bf16(
          false, a, false, bfrag, (short)0, acc[t], false, false);
    }
    __syncthreads();                                  // done reading slab kc
  }

  // Fused masked log_softmax. acc[t][v] = logits[row, col],
  // row = m_base + v + 8*(lane>=16), col = t*16 + (lane&15).
  const int mh  = (lane >> 4) * 8;
  const int bI  = m_base >> 11;
  const int len = lengths[bI];
  const int t0  = m_base & (T_ - 1);
  #pragma unroll
  for (int v = 0; v < 8; ++v) {
    const int  row       = m_base + mh + v;
    const bool valid_row = (t0 + mh + v) < len;
    float x[NT_];
    float mx = -3.0e38f;
    #pragma unroll
    for (int t = 0; t < NT_; ++t) {
      const bool validc = (t < 7) || (lc < 8);        // cols 120..127 = padding
      float val = valid_row ? acc[t][v] : 0.0f;
      x[t] = val;
      if (validc) mx = fmaxf(mx, val);
    }
    for (int m = 1; m < 16; m <<= 1) mx = fmaxf(mx, __shfl_xor(mx, m, 32));
    float s = 0.0f;
    #pragma unroll
    for (int t = 0; t < NT_; ++t) {
      const bool validc = (t < 7) || (lc < 8);
      if (validc) s += __expf(x[t] - mx);
    }
    for (int m = 1; m < 16; m <<= 1) s += __shfl_xor(s, m, 32);
    const float lse = mx + __logf(s);
    #pragma unroll
    for (int t = 0; t < NT_; ++t) {
      const int col = t * 16 + lc;
      if (col < L_) out[(size_t)row * L_ + col] = x[t] - lse;
    }
  }
}

// ---------------------------------------------------------------------------
// Kernel 2: recurrent decoder replay. One WG (512 thr = 16 wave32) per batch
// element; thread u owns LSTM unit u. fp16 weights stream from L2 (half the
// bytes of f32 -> half the per-WGP L2-bandwidth cost that dominates here).
// ---------------------------------------------------------------------------
__global__ __launch_bounds__(512) void decoder_kernel(
    const float* __restrict__ enc, const int* __restrict__ lengths,
    const int* __restrict__ boundary, const int* __restrict__ gold_pos,
    const float* __restrict__ pos_emb, const _Float16* __restrict__ wComb,
    const float* __restrict__ b_comb, const _Float16* __restrict__ wIH,
    const _Float16* __restrict__ wHH, const float* __restrict__ b_ih,
    const float* __restrict__ b_hh, float* __restrict__ out) {
  __shared__ float sh_h[HR_], sh_c[HR_], sh_S[D_], sh_Sws[D_];
  __shared__ float sh_vec[CIN_], sh_inp[HIN_];
  __shared__ int sh_ws, sh_pid;

  const int b   = blockIdx.x;
  const int tid = threadIdx.x;
  const int len = lengths[b];
  const float* enc_b = enc + (size_t)b * T_ * D_;
  const int*   bd_b  = boundary + (size_t)b * T_;
  const int*   gp_b  = gold_pos + (size_t)b * T_;

  for (int j = tid; j < D_; j += 512) { sh_S[j] = 0.0f; sh_Sws[j] = 0.0f; }
  // Initial LSTM step: x=0, h=0, c=0 -> gates are pure biases.
  if (tid < HR_) {
    const int u = tid;
    float gi = b_ih[u]         + b_hh[u];
    float gg = b_ih[2*HR_ + u] + b_hh[2*HR_ + u];
    float go = b_ih[3*HR_ + u] + b_hh[3*HR_ + u];
    float cn = sigf(gi) * tanhf(gg);
    sh_c[u] = cn;
    sh_h[u] = sigf(go) * tanhf(cn);
  }
  if (tid == 0) { sh_ws = 0; sh_pid = 0; }
  __syncthreads();

  for (int t = 0; t < T_; ++t) {
    const int  ws      = sh_ws;
    const int  pid     = sh_pid;
    const bool active  = (t < len);
    const bool do_lstm = (t >= 1) && active;           // block-uniform
    if (do_lstm) {
      // vec = [pos_emb[pid] ; (csum[t]-csum[ws]) / max(t-ws,1)]
      const float inv = 1.0f / fmaxf((float)(t - ws), 1.0f);
      for (int j = tid; j < CIN_; j += 512)
        sh_vec[j] = (j < PD_) ? pos_emb[(size_t)pid * PD_ + j]
                              : (sh_S[j - PD_] - sh_Sws[j - PD_]) * inv;
      __syncthreads();
      // inp = tanh(W_comb @ vec + b_comb): one row per thread
      {
        const int r = tid;
        const h8*     w  = (const h8*)(wComb + (size_t)r * CIN_);
        const float4* xv = (const float4*)sh_vec;
        float acc = b_comb[r];
        #pragma unroll 2
        for (int k = 0; k < CIN_ / 8; ++k) {
          h8 wv = w[k];
          float4 va = xv[2 * k], vb = xv[2 * k + 1];
          acc += (float)wv[0]*va.x + (float)wv[1]*va.y + (float)wv[2]*va.z + (float)wv[3]*va.w
               + (float)wv[4]*vb.x + (float)wv[5]*vb.y + (float)wv[6]*vb.z + (float)wv[7]*vb.w;
        }
        sh_inp[r] = tanhf(acc);
      }
      __syncthreads();
      // LSTM gates (i,f,g,o) for unit u = tid.
      const int u = tid;
      float aI = b_ih[u]         + b_hh[u];
      float aF = b_ih[HR_ + u]   + b_hh[HR_ + u];
      float aG = b_ih[2*HR_ + u] + b_hh[2*HR_ + u];
      float aO = b_ih[3*HR_ + u] + b_hh[3*HR_ + u];
      {
        const h8* wI = (const h8*)(wIH + (size_t)u * HIN_);
        const h8* wF = (const h8*)(wIH + (size_t)(HR_ + u) * HIN_);
        const h8* wG = (const h8*)(wIH + (size_t)(2*HR_ + u) * HIN_);
        const h8* wO = (const h8*)(wIH + (size_t)(3*HR_ + u) * HIN_);
        const float4* xv = (const float4*)sh_inp;
        #pragma unroll 2
        for (int k = 0; k < HIN_ / 8; ++k) {
          float4 va = xv[2 * k], vb = xv[2 * k + 1];
          h8 i8 = wI[k], f8 = wF[k], g8 = wG[k], o8 = wO[k];
          aI += (float)i8[0]*va.x + (float)i8[1]*va.y + (float)i8[2]*va.z + (float)i8[3]*va.w
              + (float)i8[4]*vb.x + (float)i8[5]*vb.y + (float)i8[6]*vb.z + (float)i8[7]*vb.w;
          aF += (float)f8[0]*va.x + (float)f8[1]*va.y + (float)f8[2]*va.z + (float)f8[3]*va.w
              + (float)f8[4]*vb.x + (float)f8[5]*vb.y + (float)f8[6]*vb.z + (float)f8[7]*vb.w;
          aG += (float)g8[0]*va.x + (float)g8[1]*va.y + (float)g8[2]*va.z + (float)g8[3]*va.w
              + (float)g8[4]*vb.x + (float)g8[5]*vb.y + (float)g8[6]*vb.z + (float)g8[7]*vb.w;
          aO += (float)o8[0]*va.x + (float)o8[1]*va.y + (float)o8[2]*va.z + (float)o8[3]*va.w
              + (float)o8[4]*vb.x + (float)o8[5]*vb.y + (float)o8[6]*vb.z + (float)o8[7]*vb.w;
        }
      }
      {
        const h8* wI = (const h8*)(wHH + (size_t)u * HR_);
        const h8* wF = (const h8*)(wHH + (size_t)(HR_ + u) * HR_);
        const h8* wG = (const h8*)(wHH + (size_t)(2*HR_ + u) * HR_);
        const h8* wO = (const h8*)(wHH + (size_t)(3*HR_ + u) * HR_);
        const float4* hv = (const float4*)sh_h;
        #pragma unroll 2
        for (int k = 0; k < HR_ / 8; ++k) {
          float4 va = hv[2 * k], vb = hv[2 * k + 1];
          h8 i8 = wI[k], f8 = wF[k], g8 = wG[k], o8 = wO[k];
          aI += (float)i8[0]*va.x + (float)i8[1]*va.y + (float)i8[2]*va.z + (float)i8[3]*va.w
              + (float)i8[4]*vb.x + (float)i8[5]*vb.y + (float)i8[6]*vb.z + (float)i8[7]*vb.w;
          aF += (float)f8[0]*va.x + (float)f8[1]*va.y + (float)f8[2]*va.z + (float)f8[3]*va.w
              + (float)f8[4]*vb.x + (float)f8[5]*vb.y + (float)f8[6]*vb.z + (float)f8[7]*vb.w;
          aG += (float)g8[0]*va.x + (float)g8[1]*va.y + (float)g8[2]*va.z + (float)g8[3]*va.w
              + (float)g8[4]*vb.x + (float)g8[5]*vb.y + (float)g8[6]*vb.z + (float)g8[7]*vb.w;
          aO += (float)o8[0]*va.x + (float)o8[1]*va.y + (float)o8[2]*va.z + (float)o8[3]*va.w
              + (float)o8[4]*vb.x + (float)o8[5]*vb.y + (float)o8[6]*vb.z + (float)o8[7]*vb.w;
        }
      }
      const float cn = sigf(aF) * sh_c[u] + sigf(aI) * tanhf(aG);
      const float hn = sigf(aO) * tanhf(cn);
      __syncthreads();                    // all sh_h reads complete
      sh_c[u] = cn;
      sh_h[u] = hn;
    }
    __syncthreads();
    // Gold-action bookkeeping; snapshot Sws BEFORE S absorbs enc[t].
    const bool isb = active && (bd_b[t] == 1);
    if (isb) for (int j = tid; j < D_; j += 512) sh_Sws[j] = sh_S[j];
    if (tid == 0) {
      if (isb)                   { sh_ws = t; sh_pid = gp_b[t]; }
      else if (t == 0 && active) { sh_pid = 0; }   // POS_UNK_ID
    }
    const float* er = enc_b + (size_t)t * D_;
    for (int j = tid; j < D_; j += 512) sh_S[j] += er[j];
    __syncthreads();
  }

  const size_t OFF_H = (size_t)B_ * T_ * L_;
  const size_t OFF_C = OFF_H + (size_t)B_ * HR_;
  if (tid < HR_) {
    out[OFF_H + (size_t)b * HR_ + tid] = sh_h[tid];
    out[OFF_C + (size_t)b * HR_ + tid] = sh_c[tid];
  }
}

// ---------------------------------------------------------------------------
extern "C" void kernel_launch(void* const* d_in, const int* in_sizes, int n_in,
                              void* d_out, int out_size, void* d_ws, size_t ws_size,
                              hipStream_t stream) {
  (void)in_sizes; (void)n_in; (void)out_size; (void)ws_size;
  const float* enc      = (const float*)d_in[0];
  const int*   lengths  = (const int*)  d_in[1];
  const int*   boundary = (const int*)  d_in[2];
  const int*   gold     = (const int*)  d_in[3];
  const float* pos_emb  = (const float*)d_in[4];
  const float* W_lin    = (const float*)d_in[5];
  const float* W_comb   = (const float*)d_in[6];
  const float* b_comb   = (const float*)d_in[7];
  const float* W_ih     = (const float*)d_in[8];
  const float* W_hh     = (const float*)d_in[9];
  const float* b_ih     = (const float*)d_in[10];
  const float* b_hh     = (const float*)d_in[11];
  float* out = (float*)d_out;

  unsigned short* ws16 = (unsigned short*)d_ws;
  __bf16*    wsB    = (__bf16*)   (ws16 + WSOFF_B);
  _Float16*  wsComb = (_Float16*) (ws16 + WSOFF_COMB);
  _Float16*  wsIH   = (_Float16*) (ws16 + WSOFF_IH);
  _Float16*  wsHH   = (_Float16*) (ws16 + WSOFF_HH);

  pack_wlin_kernel<<<32, 256, 0, stream>>>(W_lin, wsB);
  pack_f16_kernel<<<512, 256, 0, stream>>>(W_comb, wsComb, HIN_ * CIN_);
  pack_f16_kernel<<<512, 256, 0, stream>>>(W_ih, wsIH, 4 * HR_ * HIN_);
  pack_f16_kernel<<<512, 256, 0, stream>>>(W_hh, wsHH, 4 * HR_ * HR_);

  logits_lsm_kernel<<<(B_ * T_) / 64, 128, 0, stream>>>(enc, wsB, lengths, out);
  decoder_kernel<<<B_, 512, 0, stream>>>(enc, lengths, boundary, gold, pos_emb,
                                         wsComb, b_comb, wsIH, wsHH, b_ih, b_hh,
                                         out);
}
